// Based_56719338111472
// MI455X (gfx1250) — matrix operand order repaired and checked by reading, not compiled, not run on hardware
//
#include <hip/hip_runtime.h>

typedef __bf16 bf16;
typedef bf16  v16bf __attribute__((ext_vector_type(16)));
typedef bf16  v8bf  __attribute__((ext_vector_type(8)));
typedef float v8f   __attribute__((ext_vector_type(8)));
typedef float v4f   __attribute__((ext_vector_type(4)));

#define L_SEQ 512
#define DMODEL 768
#define NH 12
#define FDIM 16
#define HDIM 64
#define DPHI 273
#define DPAD 288      // 273 padded to multiple of 32
#define CH 32         // chunk length
#define NCH 16        // 512 / 32
#define FV 80         // 64 v-dims + 1 ones-column, padded to 5 N-tiles
#define INV_RRD 0.5f                    // 1 / 16^0.25
#define INV_R2RD 0.17677669529663687f   // 1 / (sqrt(2)*sqrt(16))

#if __has_builtin(__builtin_amdgcn_global_load_async_to_lds_b128) && \
    __has_builtin(__builtin_amdgcn_s_wait_asynccnt)
#define HAVE_ASYNC_LDS 1
#else
#define HAVE_ASYNC_LDS 0
#endif

// The async-to-LDS builtin wants typed 16-byte vector pointers:
//   param0: int __vector(4) addrspace(1)* (global), param1: addrspace(3)* (LDS)
typedef int v4i_gnu __attribute__((vector_size(16)));
typedef __attribute__((address_space(1))) v4i_gnu* g_v4i_ptr;
typedef __attribute__((address_space(3))) v4i_gnu* l_v4i_ptr;

// 16-byte global -> LDS copy; async (ASYNCcnt-tracked, no VGPR round trip) when available.
// Generic shared-pointer low 32 bits == LDS byte offset (aperture lives in addr[63:32]).
__device__ __forceinline__ void copy16_g2l(const float* g, float* l) {
#if HAVE_ASYNC_LDS
  __builtin_amdgcn_global_load_async_to_lds_b128(
      (g_v4i_ptr)(uintptr_t)g, (l_v4i_ptr)(uint32_t)(uintptr_t)l, 0, 0);
#else
  *(v4f*)l = *(const v4f*)g;
#endif
}
__device__ __forceinline__ void wait_async_lds() {
#if HAVE_ASYNC_LDS
  __builtin_amdgcn_s_wait_asynccnt(0);
#endif
}

// ---- CDNA5 wave32 WMMA fragment loaders (16x16x32 bf16) ----
// A (16x32, MxK): lane L holds row m0+(L&15); elements 0..7 -> K = k0 + (L>=16)*8 + e,
// elements 8..15 -> K = k0+16 + (L>=16)*8 + (e-8). Two contiguous 16B loads.
__device__ __forceinline__ v16bf load_a_frag(const bf16* src, int ld, int m0, int k0, int lane) {
  int m   = m0 + (lane & 15);
  int kh8 = (lane >> 4) << 3;
  const bf16* p = src + m * ld + k0 + kh8;
  v8bf lo = *(const v8bf*)p;
  v8bf hi = *(const v8bf*)(p + 16);
  return __builtin_shufflevector(lo, hi, 0,1,2,3,4,5,6,7,8,9,10,11,12,13,14,15);
}
// B (32x16, KxN): lane L holds col n0+(L&15); elements 0..7 -> K = k0 + (L>=16)*8 + e,
// elements 8..15 -> K = k0+16 + (L>=16)*8 + (e-8). src: B[k][n] = src[n*ld + k] -> two 16B loads.
__device__ __forceinline__ v16bf load_b_frag(const bf16* src, int ld, int n0, int k0, int lane) {
  int n = n0 + (lane & 15);
  const bf16* p = src + n * ld + k0 + ((lane >> 4) << 3);
  v8bf lo = *(const v8bf*)p;
  v8bf hi = *(const v8bf*)(p + 16);
  return __builtin_shufflevector(lo, hi, 0,1,2,3,4,5,6,7,8,9,10,11,12,13,14,15);
}

__global__ void cvt_f32_bf16(const float* __restrict__ in, bf16* __restrict__ out, int n) {
  int i = blockIdx.x * blockDim.x + threadIdx.x;
  if (i < n) out[i] = (bf16)in[i];
}

// C[M][N] (f32, row-major) = A[M][K] (bf16, row-major) @ B, where B[k][n] = Bw[n*K + k]
// (i.e. Bw is the original nn.Linear weight [N][K] converted to bf16). One wave per tile.
__global__ __launch_bounds__(128) void wmma_gemm_bf16(
    const bf16* __restrict__ A, const bf16* __restrict__ Bw,
    float* __restrict__ C, int M, int N, int K) {
  int wave = blockIdx.x * (blockDim.x >> 5) + (threadIdx.x >> 5);
  int lane = threadIdx.x & 31;
  int ntiles = N >> 4;
  int mt = wave / ntiles, nt = wave - mt * ntiles;
  if (mt >= (M >> 4)) return;
  v8f acc = {};
  for (int k0 = 0; k0 < K; k0 += 32) {
    v16bf a = load_a_frag(A,  K, mt * 16, k0, lane);
    v16bf b = load_b_frag(Bw, K, nt * 16, k0, lane);
    acc = __builtin_amdgcn_wmma_f32_16x16x32_bf16(false, a, false, b, (short)0, acc, false, false);
  }
  int row0 = mt * 16 + ((lane >> 4) << 3);
  int col  = nt * 16 + (lane & 15);
#pragma unroll
  for (int r = 0; r < 8; ++r)
    C[(row0 + r) * N + col] = acc[r];
}

// One workgroup per head: chunked causal linear attention with Taylor feature map.
// State S[288][80] kept as WMMA f32 accumulators in registers (9 tiles per wave x 10 waves),
// mirrored to LDS in bf16 (transposed: SbT[n][k]) for the inter-chunk B-fragments.
// q/k/v chunks are double-buffered in LDS via async global->LDS copies (prefetched one
// chunk ahead, overlapped with the WMMA pipeline of the current chunk).
__global__ __launch_bounds__(320, 1) void based_attn_head(
    const float* __restrict__ qf, const float* __restrict__ kf,
    const float* __restrict__ vf, bf16* __restrict__ Yb) {
  const int h    = blockIdx.x;
  const int tid  = threadIdx.x;
  const int nthr = 320;
  const int w    = tid >> 5;
  const int lane = tid & 31;
  const int kh8  = (lane >> 4) << 3;

  __shared__ alignas(16) float qx2[2][CH][FDIM];   // raw q chunk (double-buffered)
  __shared__ alignas(16) float kx2[2][CH][FDIM];   // raw k chunk
  __shared__ alignas(16) float Vs2[2][CH][HDIM];   // raw v chunk
  __shared__ alignas(32) bf16  PhiQ[CH][DPAD];     // [pos][feature]
  __shared__ alignas(32) bf16  PhiK[CH][DPAD];     // [pos][feature]
  __shared__ alignas(32) bf16  PhiKT[DPAD][CH];    // [feature][pos]
  __shared__ alignas(32) bf16  VcT[FV][CH];        // [vdim(+ones)][pos]
  __shared__ alignas(32) bf16  Amask[CH][CH];      // causal-masked scores
  __shared__ alignas(32) bf16  SbT[FV][DPAD];      // bf16 shadow of state, SbT[n][k]=S[k][n]
  __shared__ alignas(32) float Yf[CH][FV];

  // Issue async loads of chunk cc into buffer bb (q/k: 128 x b128 each, v: 512 x b128).
  auto issue_loads = [&](int cc, int bb) {
    const int l0 = cc * CH;
    if (tid < 256) {
      int t  = tid & 127;
      int r  = t >> 2, j4 = (t & 3) << 2;
      if (tid < 128)
        copy16_g2l(qf + (l0 + r) * (NH * FDIM) + h * FDIM + j4, &qx2[bb][r][j4]);
      else
        copy16_g2l(kf + (l0 + r) * (NH * FDIM) + h * FDIM + j4, &kx2[bb][r][j4]);
    }
    for (int t = tid; t < CH * (HDIM / 4); t += nthr) {
      int r = t >> 4, j4 = (t & 15) << 2;
      copy16_g2l(vf + (l0 + r) * DMODEL + h * HDIM + j4, &Vs2[bb][r][j4]);
    }
  };

  v8f zero = {};
  v8f sacc[9];
#pragma unroll
  for (int i = 0; i < 9; ++i) sacc[i] = zero;

  for (int i = tid; i < FV * DPAD; i += nthr) (&SbT[0][0])[i] = (bf16)0.0f;

  const int ymt = w / 5, ynt = w % 5;   // this wave's Y tile (2 x 5 tiles of 16x16)

  issue_loads(0, 0);
  int buf = 0;

  for (int c = 0; c < NCH; ++c) {
    const int l0 = c * CH;
    wait_async_lds();        // this wave's async copies for buffer `buf` have landed
    __syncthreads();         // ... and every wave's copies are visible

    // prefetch next chunk into the other buffer while we compute on this one
    if (c + 1 < NCH) issue_loads(c + 1, buf ^ 1);

    // --- Vext (v columns + ones column) transposed to bf16 ---
    for (int i = tid; i < FV * CH; i += nthr) {
      int n = i >> 5, s = i & 31;
      float val = (n < HDIM) ? Vs2[buf][s][n] : (n == HDIM ? 1.0f : 0.0f);
      VcT[n][s] = (bf16)val;
    }
    // --- Taylor feature map phi = [1, x/d^0.25, (x (x) x)/(sqrt2*sqrt d)] ---
    for (int i = tid; i < CH * DPAD; i += nthr) {
      int r = i / DPAD, col = i - r * DPAD;
      float pq, pk;
      if (col == 0)          { pq = 1.0f; pk = 1.0f; }
      else if (col < 17)     { pq = qx2[buf][r][col - 1] * INV_RRD;
                               pk = kx2[buf][r][col - 1] * INV_RRD; }
      else if (col < DPHI)   { int a = (col - 17) >> 4, b = (col - 17) & 15;
                               pq = qx2[buf][r][a] * qx2[buf][r][b] * INV_R2RD;
                               pk = kx2[buf][r][a] * kx2[buf][r][b] * INV_R2RD; }
      else                   { pq = 0.0f; pk = 0.0f; }
      PhiQ[r][col]  = (bf16)pq;
      PhiK[r][col]  = (bf16)pk;
      PhiKT[col][r] = (bf16)pk;
    }
    __syncthreads();

    // --- inter-chunk: Y = PhiQ @ S_prev  (all 10 waves, 9 K-iters each) ---
    v8f yacc = zero;
#pragma unroll
    for (int k0 = 0; k0 < DPAD; k0 += 32) {
      v16bf a = load_a_frag(&PhiQ[0][0], DPAD, ymt * 16, k0, lane);
      v16bf b = load_b_frag(&SbT[0][0],  DPAD, ynt * 16, k0, lane);
      yacc = __builtin_amdgcn_wmma_f32_16x16x32_bf16(false, a, false, b, (short)0, yacc, false, false);
    }

    // --- intra-chunk scores A = PhiQ @ PhiK^T on waves 0..3 (2x2 tiles), masked ---
    if (w < 4) {
      int amt = w >> 1, ant = w & 1;
      v8f aacc = zero;
#pragma unroll
      for (int k0 = 0; k0 < DPAD; k0 += 32) {
        v16bf a = load_a_frag(&PhiQ[0][0], DPAD, amt * 16, k0, lane);
        v16bf b = load_b_frag(&PhiK[0][0], DPAD, ant * 16, k0, lane);
        aacc = __builtin_amdgcn_wmma_f32_16x16x32_bf16(false, a, false, b, (short)0, aacc, false, false);
      }
#pragma unroll
      for (int r = 0; r < 8; ++r) {
        int row = amt * 16 + r + kh8;       // t (query pos in chunk)
        int col = ant * 16 + (lane & 15);   // s (key pos in chunk)
        Amask[row][col] = (bf16)((col <= row) ? aacc[r] : 0.0f);
      }
    }
    __syncthreads();

    // --- Y += masked(A) @ Vext  (single K=32 WMMA per tile) ---
    {
      v16bf a = load_a_frag(&Amask[0][0], CH, ymt * 16, 0, lane);
      v16bf b = load_b_frag(&VcT[0][0],   CH, ynt * 16, 0, lane);
      yacc = __builtin_amdgcn_wmma_f32_16x16x32_bf16(false, a, false, b, (short)0, yacc, false, false);
    }
#pragma unroll
    for (int r = 0; r < 8; ++r)
      Yf[ymt * 16 + r + kh8][ynt * 16 + (lane & 15)] = yacc[r];

    // --- state update: S += PhiK^T @ Vext  (9 register-resident tiles per wave) ---
#pragma unroll
    for (int tIdx = 0; tIdx < 9; ++tIdx) {
      int idx = w * 9 + tIdx;            // 0..89 over 18x5 tiles
      int smt = idx / 5, snt = idx - smt * 5;
      v16bf a = load_a_frag(&PhiKT[0][0], CH, smt * 16, 0, lane);
      v16bf b = load_b_frag(&VcT[0][0],   CH, snt * 16, 0, lane);
      sacc[tIdx] = __builtin_amdgcn_wmma_f32_16x16x32_bf16(false, a, false, b, (short)0, sacc[tIdx], false, false);
    }
    // mirror updated state to LDS (bf16, transposed) for next chunk's B-fragments
#pragma unroll
    for (int tIdx = 0; tIdx < 9; ++tIdx) {
      int idx = w * 9 + tIdx;
      int smt = idx / 5, snt = idx - smt * 5;
#pragma unroll
      for (int r = 0; r < 8; ++r) {
        int mrow = smt * 16 + r + kh8;        // feature dim (0..287)
        int ncol = snt * 16 + (lane & 15);    // v dim (0..79)
        SbT[ncol][mrow] = (bf16)sacc[tIdx][r];
      }
    }
    __syncthreads();

    // --- y = num / (den + eps), emit bf16 for the output projection ---
    for (int i = tid; i < CH * HDIM; i += nthr) {
      int s = i >> 6, j = i & 63;
      float den = Yf[s][HDIM] + 1e-12f;
      Yb[(l0 + s) * DMODEL + h * HDIM + j] = (bf16)(Yf[s][j] / den);
    }
    __syncthreads();
    buf ^= 1;
  }
}

extern "C" void kernel_launch(void* const* d_in, const int* in_sizes, int n_in,
                              void* d_out, int out_size, void* d_ws, size_t ws_size,
                              hipStream_t stream) {
  (void)in_sizes; (void)n_in; (void)out_size; (void)ws_size;
  const float* hs = (const float*)d_in[0];
  const float* Wq = (const float*)d_in[1];
  const float* Wk = (const float*)d_in[2];
  const float* Wv = (const float*)d_in[3];
  const float* Wo = (const float*)d_in[4];
  float* out = (float*)d_out;

  char* p = (char*)d_ws;
  auto alloc = [&](size_t bytes) -> char* {
    char* r = p; p += (bytes + 255) & ~(size_t)255; return r;
  };
  bf16*  hsb = (bf16*)alloc((size_t)L_SEQ * DMODEL * 2);
  bf16*  Wqb = (bf16*)alloc((size_t)(NH * FDIM) * DMODEL * 2);
  bf16*  Wkb = (bf16*)alloc((size_t)(NH * FDIM) * DMODEL * 2);
  bf16*  Wvb = (bf16*)alloc((size_t)DMODEL * DMODEL * 2);
  bf16*  Wob = (bf16*)alloc((size_t)DMODEL * DMODEL * 2);
  float* qfp = (float*)alloc((size_t)L_SEQ * (NH * FDIM) * 4);
  float* kfp = (float*)alloc((size_t)L_SEQ * (NH * FDIM) * 4);
  float* vfp = (float*)alloc((size_t)L_SEQ * DMODEL * 4);
  bf16*  Ybf = (bf16*)alloc((size_t)L_SEQ * DMODEL * 2);

  const int T = 256;
  cvt_f32_bf16<<<(L_SEQ * DMODEL + T - 1) / T, T, 0, stream>>>(hs, hsb, L_SEQ * DMODEL);
  cvt_f32_bf16<<<(NH * FDIM * DMODEL + T - 1) / T, T, 0, stream>>>(Wq, Wqb, NH * FDIM * DMODEL);
  cvt_f32_bf16<<<(NH * FDIM * DMODEL + T - 1) / T, T, 0, stream>>>(Wk, Wkb, NH * FDIM * DMODEL);
  cvt_f32_bf16<<<(DMODEL * DMODEL + T - 1) / T, T, 0, stream>>>(Wv, Wvb, DMODEL * DMODEL);
  cvt_f32_bf16<<<(DMODEL * DMODEL + T - 1) / T, T, 0, stream>>>(Wo, Wob, DMODEL * DMODEL);

  // projections: q,k -> [512][192], v -> [512][768]
  wmma_gemm_bf16<<<(32 * 12) / 4, 128, 0, stream>>>(hsb, Wqb, qfp, L_SEQ, NH * FDIM, DMODEL);
  wmma_gemm_bf16<<<(32 * 12) / 4, 128, 0, stream>>>(hsb, Wkb, kfp, L_SEQ, NH * FDIM, DMODEL);
  wmma_gemm_bf16<<<(32 * 48) / 4, 128, 0, stream>>>(hsb, Wvb, vfp, L_SEQ, DMODEL, DMODEL);

  based_attn_head<<<NH, 320, 0, stream>>>(qfp, kfp, vfp, Ybf);

  // output projection: out = Y @ Wo^T
  wmma_gemm_bf16<<<(32 * 48) / 4, 128, 0, stream>>>(Ybf, Wob, out, L_SEQ, DMODEL, DMODEL);
}
